// GuidedPrototypeAllocation_46196668236378
// MI455X (gfx1250) — compile-verified, matrix operand order checked by reading, not run on hardware
//
#include <hip/hip_runtime.h>

typedef _Float16 v16h __attribute__((ext_vector_type(16)));
typedef _Float16 h8   __attribute__((ext_vector_type(8)));
typedef float    v8f  __attribute__((ext_vector_type(8)));

#define B_ 4
#define C_ 256
#define H_ 128
#define W_ 128
#define S_ (H_ * W_)
#define N_ 512
#define INV_TEMP 10.0f

// --- WMMA helpers -----------------------------------------------------------
// A (16x32, f16) / B(32x16 stored transposed [n][k]) fragment, row-major src:
// element [r][k] lives at base[r*stride + k].  Per ISA 7.12.2, lane L holds
// r = L&15; halves 0..7 -> k = k0 + (L>=16?8:0) + 0..7, halves 8..15 -> +16.
__device__ __forceinline__ v16h load_frag16(const _Float16* base, int stride,
                                            int k0, int lane) {
  int r = lane & 15;
  int ks = k0 + ((lane >> 4) << 3);
  const _Float16* p = base + (size_t)r * stride + ks;
  h8 lo = *(const h8*)(p);
  h8 hi = *(const h8*)(p + 16);
  v16h f;
#pragma unroll
  for (int i = 0; i < 8; ++i) { f[i] = lo[i]; f[i + 8] = hi[i]; }
  return f;
}

__device__ __forceinline__ v8f wmma_f16(v16h a, v16h b, v8f c) {
  return __builtin_amdgcn_wmma_f32_16x16x32_f16(false, a, false, b,
                                                (short)0, c, false, false);
}

// Async global->LDS copy, 16B per lane, tracked on ASYNCcnt.
__device__ __forceinline__ void async_copy_b128(const _Float16* lds_dst,
                                                const _Float16* gsrc) {
  unsigned lds_off = (unsigned)(uintptr_t)lds_dst;  // low 32 bits = LDS offset
  unsigned long long ga = (unsigned long long)(uintptr_t)gsrc;
  asm volatile("global_load_async_to_lds_b128 %0, %1, off"
               :: "v"(lds_off), "v"(ga)
               : "memory");
}

__device__ __forceinline__ void wait_asynccnt0() {
  asm volatile("s_wait_asynccnt 0x0" ::: "memory");
}

// --- K0: prototype prep -----------------------------------------------------
__global__ void proto_prep_kernel(const float* __restrict__ proto,
                                  _Float16* __restrict__ pn,
                                  _Float16* __restrict__ pT) {
  __shared__ float red[256];
  int n = blockIdx.x, c = threadIdx.x;
  float v = proto[n * C_ + c];
  red[c] = v * v;
  __syncthreads();
  for (int off = 128; off > 0; off >>= 1) {
    if (c < off) red[c] += red[c + off];
    __syncthreads();
  }
  float rinv = INV_TEMP / fmaxf(sqrtf(red[0]), 1e-12f);
  pn[n * C_ + c] = (_Float16)(v * rinv);
  pT[c * N_ + n] = (_Float16)v;
}

// --- K0b: conv weight repack to (co, tap, ci) f16 ---------------------------
__global__ void conv_pack_kernel(const float* __restrict__ w,
                                 _Float16* __restrict__ wp) {
  int co = blockIdx.x, ci = threadIdx.x;
#pragma unroll
  for (int tap = 0; tap < 9; ++tap) {
    int ky = tap / 3, kx = tap % 3;
    wp[co * 2304 + tap * 256 + ci] =
        (_Float16)w[((co * 256 + ci) * 3 + ky) * 3 + kx];
  }
}

// --- K1: fused q-norm -> corr GEMM -> softmax -> allocate GEMM --------------
// One block = one batch x 32 spatial columns.  8 waves.
// Output `allocated` is emitted as f16, channel-last [b][sp][c] so the conv
// can stage it with async B128 copies.
__global__ void __launch_bounds__(256, 1)
assign_kernel(const float* __restrict__ q, const _Float16* __restrict__ pn,
              const _Float16* __restrict__ pT, _Float16* __restrict__ alloc_h) {
  __shared__ __align__(16) _Float16 sQ[32 * 256];   // [col][c]  16 KB
  __shared__ __align__(16) float    sC[512 * 32];   // [n][col]  64 KB
  __shared__ __align__(16) _Float16 sW[32 * 512];   // [col][n]  32 KB
  __shared__ float sRed[8 * 32];
  __shared__ float sStat[32];

  int t = threadIdx.x;
  int lane = t & 31, wave = t >> 5;
  int b = blockIdx.x >> 9;
  int s0 = (blockIdx.x & 511) << 5;

  float* sTmp = sC;  // reuse corr buffer as [c][col] f32 q staging

  // phase 0: stage q tile, compute per-column inverse L2 norm
  const float* qb = q + (size_t)b * C_ * S_ + s0;
  for (int idx = t; idx < 256 * 32; idx += 256) {
    int c = idx >> 5, col = idx & 31;
    sTmp[c * 32 + col] = qb[(size_t)c * S_ + col];
  }
  __syncthreads();
  {
    int col = t & 31, part = t >> 5;
    float s = 0.f;
    for (int c = part * 32; c < part * 32 + 32; ++c) {
      float v = sTmp[c * 32 + col];
      s += v * v;
    }
    sRed[part * 32 + col] = s;
  }
  __syncthreads();
  if (t < 32) {
    float s = 0.f;
    for (int p = 0; p < 8; ++p) s += sRed[p * 32 + t];
    sStat[t] = 1.0f / fmaxf(sqrtf(s), 1e-12f);
  }
  __syncthreads();
  for (int idx = t; idx < 256 * 32; idx += 256) {
    int c = idx >> 5, col = idx & 31;
    sQ[col * 256 + c] = (_Float16)(sTmp[c * 32 + col] * sStat[col]);
  }
  __syncthreads();

  // phase 1: corr(512x32) = pn(512x256) x qn(256x32), WMMA
  v8f acc1[4][2] = {};
  int n0 = wave * 64;
#pragma unroll 1
  for (int k0 = 0; k0 < 256; k0 += 32) {
    v16h a[4], bb[2];
#pragma unroll
    for (int rt = 0; rt < 4; ++rt)
      a[rt] = load_frag16(pn + (size_t)(n0 + rt * 16) * 256, 256, k0, lane);
#pragma unroll
    for (int ct = 0; ct < 2; ++ct)
      bb[ct] = load_frag16(sQ + (ct * 16) * 256, 256, k0, lane);
#pragma unroll
    for (int rt = 0; rt < 4; ++rt)
#pragma unroll
      for (int ct = 0; ct < 2; ++ct)
        acc1[rt][ct] = wmma_f16(a[rt], bb[ct], acc1[rt][ct]);
  }
  {
    int colL = lane & 15, hi = (lane >> 4) << 3;
#pragma unroll
    for (int rt = 0; rt < 4; ++rt)
#pragma unroll
      for (int ct = 0; ct < 2; ++ct)
#pragma unroll
        for (int r = 0; r < 8; ++r)
          sC[(n0 + rt * 16 + r + hi) * 32 + ct * 16 + colL] = acc1[rt][ct][r];
  }
  __syncthreads();

  // phase 2: softmax over N=512 per column; weights -> sW[col][n] f16
  {
    int col = t & 31, seg = t >> 5;
    int r0 = seg * 64;
    float m = -3.4e38f;
    for (int r = r0; r < r0 + 64; ++r) m = fmaxf(m, sC[r * 32 + col]);
    sRed[seg * 32 + col] = m;
    __syncthreads();
    if (t < 32) {
      float mm = sRed[t];
      for (int p = 1; p < 8; ++p) mm = fmaxf(mm, sRed[p * 32 + t]);
      sStat[t] = mm;
    }
    __syncthreads();
    float gm = sStat[col];
    float s = 0.f;
    for (int r = r0; r < r0 + 64; ++r) {
      float e = __expf(sC[r * 32 + col] - gm);
      sC[r * 32 + col] = e;
      s += e;
    }
    sRed[seg * 32 + col] = s;
    __syncthreads();
    if (t < 32) {
      float ss = 0.f;
      for (int p = 0; p < 8; ++p) ss += sRed[p * 32 + t];
      sStat[t] = 1.0f / ss;
    }
    __syncthreads();
    float inv = sStat[col];
    for (int r = r0; r < r0 + 64; ++r)
      sW[col * 512 + r] = (_Float16)(sC[r * 32 + col] * inv);
  }
  __syncthreads();

  // phase 3: allocated(256x32) = pT(256x512) x W(512x32), WMMA
  v8f acc2[2][2] = {};
  int c0 = wave * 32;
#pragma unroll 1
  for (int k0 = 0; k0 < 512; k0 += 32) {
    v16h a[2], bb[2];
#pragma unroll
    for (int rt = 0; rt < 2; ++rt)
      a[rt] = load_frag16(pT + (size_t)(c0 + rt * 16) * 512, 512, k0, lane);
#pragma unroll
    for (int ct = 0; ct < 2; ++ct)
      bb[ct] = load_frag16(sW + (ct * 16) * 512, 512, k0, lane);
#pragma unroll
    for (int rt = 0; rt < 2; ++rt)
#pragma unroll
      for (int ct = 0; ct < 2; ++ct)
        acc2[rt][ct] = wmma_f16(a[rt], bb[ct], acc2[rt][ct]);
  }
  {
    int colL = lane & 15, hi = (lane >> 4) << 3;
#pragma unroll
    for (int rt = 0; rt < 2; ++rt)
#pragma unroll
      for (int ct = 0; ct < 2; ++ct) {
        h8 v;
#pragma unroll
        for (int r = 0; r < 8; ++r) v[r] = (_Float16)acc2[rt][ct][r];
        int sp = s0 + ct * 16 + colL;
        *(h8*)(alloc_h + (size_t)(b * S_ + sp) * C_ + c0 + rt * 16 + hi) = v;
      }
  }
}

// --- K2: 3x3 conv as implicit GEMM (K = 9*256), WMMA ------------------------
// One block: batch b, image row yrow, 64-pixel column strip.  Halo tile is
// staged from the channel-last f16 `allocated` with async B128 copies.
__global__ void __launch_bounds__(256, 1)
conv_kernel(const _Float16* __restrict__ x, const _Float16* __restrict__ wp,
            float* __restrict__ y) {
  __shared__ __align__(16) _Float16 sX[3 * 66 * 256];  // [dy][col][ci] ~99 KB
  int t = threadIdx.x, lane = t & 31, wave = t >> 5;
  int b = blockIdx.x >> 8;
  int rem = blockIdx.x & 255;
  int yrow = rem >> 1, x0 = (rem & 1) * 64;
  int co0 = wave * 32;

  // L2 prefetch of this wave's weight rows (reused by all 1024 blocks)
  __builtin_prefetch(wp + (size_t)(co0 + lane) * 2304, 0, 3);

  const _Float16* xb = x + (size_t)b * S_ * C_;
  // 198 halo lines, one async B128 per lane per line (512B line = 1 wave op)
  for (int pr = wave; pr < 3 * 66; pr += 8) {
    int dy = pr / 66, col = pr % 66;
    int yy = yrow + dy - 1, xx = x0 - 1 + col;
    _Float16* dst = sX + (size_t)pr * 256 + lane * 8;
    if (yy >= 0 && yy < H_ && xx >= 0 && xx < W_) {
      async_copy_b128(dst, xb + (size_t)(yy * W_ + xx) * C_ + lane * 8);
    } else {
      h8 z;
#pragma unroll
      for (int i = 0; i < 8; ++i) z[i] = (_Float16)0.f;
      *(h8*)dst = z;
    }
  }
  wait_asynccnt0();
  __syncthreads();

  v8f acc[2][4] = {};
  for (int tap = 0; tap < 9; ++tap) {
    int dy = tap / 3, dx = tap % 3;
#pragma unroll 1
    for (int kc = 0; kc < 256; kc += 32) {
      v16h a[2], bb[4];
#pragma unroll
      for (int rt = 0; rt < 2; ++rt)
        a[rt] = load_frag16(wp + (size_t)(co0 + rt * 16) * 2304 + tap * 256,
                            2304, kc, lane);
#pragma unroll
      for (int pt = 0; pt < 4; ++pt)
        bb[pt] = load_frag16(sX + (size_t)(dy * 66 + dx + pt * 16) * 256, 256,
                             kc, lane);
#pragma unroll
      for (int rt = 0; rt < 2; ++rt)
#pragma unroll
        for (int pt = 0; pt < 4; ++pt)
          acc[rt][pt] = wmma_f16(a[rt], bb[pt], acc[rt][pt]);
    }
  }
  {
    int colL = lane & 15, hi = (lane >> 4) << 3;
    float* yb = y + (size_t)b * C_ * S_ + yrow * W_ + x0;
#pragma unroll
    for (int rt = 0; rt < 2; ++rt)
#pragma unroll
      for (int pt = 0; pt < 4; ++pt)
#pragma unroll
        for (int r = 0; r < 8; ++r)
          yb[(size_t)(co0 + rt * 16 + r + hi) * S_ + pt * 16 + colL] =
              acc[rt][pt][r];
  }
}

// --- K3a: GroupNorm partial stats -> atomic f32 accumulate ------------------
__global__ void gn_stats_kernel(const float* __restrict__ y,
                                float* __restrict__ stats) {
  __shared__ float rs[256], rss[256];
  int t = threadIdx.x;
  int bg = blockIdx.x >> 5;   // (b,g)
  int chunk = blockIdx.x & 31;
  int b = bg >> 3, g = bg & 7;
  const float* base = y + (size_t)(b * 256 + g * 32) * S_ + chunk * 512;
  float s = 0.f, ss = 0.f;
  for (int i = 0; i < 64; ++i) {
    int idx = t + i * 256;  // 32ch x 512sp slab
    int ch = idx >> 9, sp = idx & 511;
    float v = base[(size_t)ch * S_ + sp];
    s += v;
    ss += v * v;
  }
  rs[t] = s;
  rss[t] = ss;
  __syncthreads();
  for (int off = 128; off > 0; off >>= 1) {
    if (t < off) { rs[t] += rs[t + off]; rss[t] += rss[t + off]; }
    __syncthreads();
  }
  if (t == 0) {
    atomicAdd(&stats[(b * 8 + g) * 2 + 0], rs[0]);
    atomicAdd(&stats[(b * 8 + g) * 2 + 1], rss[0]);
  }
}

// --- K3b: GroupNorm apply + affine + ReLU (b128 streaming) ------------------
__global__ void gn_apply_kernel(const float* __restrict__ y,
                                const float* __restrict__ stats,
                                const float* __restrict__ gw,
                                const float* __restrict__ gb,
                                float* __restrict__ out) {
  size_t i4 = ((size_t)blockIdx.x * 256 + threadIdx.x) * 4;
  int b = (int)(i4 >> 22);
  int c = (int)(i4 >> 14) & 255;
  int g = c >> 5;
  float s1 = stats[(b * 8 + g) * 2 + 0];
  float s2 = stats[(b * 8 + g) * 2 + 1];
  const float inv_cnt = 1.0f / (32.0f * 16384.0f);
  float mean = s1 * inv_cnt;
  float var = s2 * inv_cnt - mean * mean;
  float sc = rsqrtf(var + 1e-5f) * gw[c];
  float bi = gb[c] - mean * rsqrtf(var + 1e-5f) * gw[c];
  float4 v = *(const float4*)(y + i4);
  float4 o;
  o.x = fmaxf(v.x * sc + bi, 0.f);
  o.y = fmaxf(v.y * sc + bi, 0.f);
  o.z = fmaxf(v.z * sc + bi, 0.f);
  o.w = fmaxf(v.w * sc + bi, 0.f);
  *(float4*)(out + i4) = o;
}

extern "C" void kernel_launch(void* const* d_in, const int* in_sizes, int n_in,
                              void* d_out, int out_size, void* d_ws,
                              size_t ws_size, hipStream_t stream) {
  (void)in_sizes; (void)n_in; (void)out_size; (void)ws_size;
  const float* q     = (const float*)d_in[0];
  const float* proto = (const float*)d_in[1];
  const float* convw = (const float*)d_in[2];
  const float* gw    = (const float*)d_in[3];
  const float* gb    = (const float*)d_in[4];
  float* out = (float*)d_out;

  char* ws = (char*)d_ws;
  size_t off = 0;
  _Float16* alloc_h = (_Float16*)(ws + off); off += (size_t)B_ * S_ * C_ * 2;
  float* yconv = (float*)(ws + off);         off += (size_t)B_ * C_ * S_ * 4;
  _Float16* pn = (_Float16*)(ws + off);      off += (size_t)N_ * C_ * 2;
  _Float16* pT = (_Float16*)(ws + off);      off += (size_t)C_ * N_ * 2;
  _Float16* wp = (_Float16*)(ws + off);      off += (size_t)C_ * 2304 * 2;
  float* stats = (float*)(ws + off);         // 64 floats

  hipMemsetAsync(stats, 0, 64 * sizeof(float), stream);
  proto_prep_kernel<<<N_, 256, 0, stream>>>(proto, pn, pT);
  conv_pack_kernel<<<C_, 256, 0, stream>>>(convw, wp);
  assign_kernel<<<B_ * (S_ / 32), 256, 0, stream>>>(q, pn, pT, alloc_h);
  conv_kernel<<<B_ * H_ * (W_ / 64), 256, 0, stream>>>(alloc_h, wp, yconv);
  gn_stats_kernel<<<B_ * 8 * 32, 256, 0, stream>>>(yconv, stats);
  gn_apply_kernel<<<(B_ * C_ * S_) / 1024, 256, 0, stream>>>(yconv, stats, gw,
                                                             gb, out);
}
